// CMGPTClassifier_16810501996994
// MI455X (gfx1250) — matrix operand order repaired and checked by reading, not compiled, not run on hardware
//
#include <hip/hip_runtime.h>

// ---------------------------------------------------------------------------
// CDNA5 (gfx1250) GPT-mini forward.
//  - all large matmuls: v_wmma_f32_16x16x32_f16 (f16 in, f32 accum)
//  - GEMM A-tile staging: global_load_async_to_lds_b128 (ASYNCcnt path)
//  - attention: flash-style online softmax, V pre-transposed per head so
//    every WMMA fragment load is a contiguous 16B load
// ---------------------------------------------------------------------------

typedef __attribute__((ext_vector_type(16))) _Float16 v16h;
typedef __attribute__((ext_vector_type(8)))  float    v8f;

#define GB 8           // batch
#define GS 1024        // seq len
#define GD 768         // model dim
#define GH 12          // heads
#define GHS 64         // head size
#define GL 4           // layers
#define GDF 3072       // ffn dim
#define GNC 16         // classes

// ============================ elementwise kernels ==========================

__global__ __launch_bounds__(256) void embed_k(const int* __restrict__ x,
                                               const float* __restrict__ tok,
                                               const float* __restrict__ pos,
                                               float* __restrict__ h) {
  long row = blockIdx.x;              // b*S + s
  int  s   = (int)(row % GS);
  long id  = x[row];
  const float* t = tok + id * (long)GD;
  const float* p = pos + (long)s * GD;
  float* o = h + row * GD;
  for (int i = threadIdx.x; i < GD; i += 256) o[i] = t[i] + p[i];
}

__global__ __launch_bounds__(256) void cvt_k(const float* __restrict__ in,
                                             _Float16* __restrict__ out, long n) {
  for (long i = blockIdx.x * 256L + threadIdx.x; i < n; i += gridDim.x * 256L)
    out[i] = (_Float16)in[i];
}

// (L,H,D,HS) -> (L, D, H*HS) row-major, f32 -> f16
__global__ __launch_bounds__(256) void repack_qkv_k(const float* __restrict__ in,
                                                    _Float16* __restrict__ out, long n) {
  for (long i = blockIdx.x * 256L + threadIdx.x; i < n; i += gridDim.x * 256L) {
    long l   = i / ((long)GD * GD);
    long rem = i % ((long)GD * GD);
    long d   = rem / GD;
    long c   = rem % GD;
    long hh  = c / GHS;
    long e   = c % GHS;
    out[i] = (_Float16)in[((l * GH + hh) * GD + d) * GHS + e];
  }
}

// v: [B*S, D] (col = h*64+e)  ->  vt: [(b*H + h)*HS + e, S]
__global__ __launch_bounds__(256) void vtrans_k(const _Float16* __restrict__ v,
                                                _Float16* __restrict__ vt) {
  const long n = (long)GB * GS * GD;
  for (long i = blockIdx.x * 256L + threadIdx.x; i < n; i += gridDim.x * 256L) {
    long row = i / GD;          // b*S + s
    int  c   = (int)(i % GD);   // h*64+e
    int  b   = (int)(row / GS);
    int  s   = (int)(row % GS);
    vt[(((long)b * GH + (c >> 6)) * GHS + (c & 63)) * GS + s] = v[i];
  }
}

// ============================ layernorm ====================================

__global__ __launch_bounds__(256) void ln_k(const float* __restrict__ in,
                                            const float* __restrict__ w,
                                            const float* __restrict__ bp,
                                            _Float16* __restrict__ out) {
  __shared__ float red[256];
  long row = blockIdx.x;
  const float* xr = in + row * GD;
  int tid = threadIdx.x;

  float s = 0.f;
  for (int i = tid; i < GD; i += 256) s += xr[i];
  red[tid] = s; __syncthreads();
  for (int o = 128; o > 0; o >>= 1) { if (tid < o) red[tid] += red[tid + o]; __syncthreads(); }
  float mu = red[0] / (float)GD;
  __syncthreads();

  float v = 0.f;
  for (int i = tid; i < GD; i += 256) { float d = xr[i] - mu; v += d * d; }
  red[tid] = v; __syncthreads();
  for (int o = 128; o > 0; o >>= 1) { if (tid < o) red[tid] += red[tid + o]; __syncthreads(); }
  float rstd = rsqrtf(red[0] / (float)GD + 1e-5f);

  _Float16* orow = out + row * GD;
  for (int i = tid; i < GD; i += 256)
    orow[i] = (_Float16)((xr[i] - mu) * rstd * w[i] + bp[i]);
}

// final LN applied to the last token of each batch row, fp32 out
__global__ __launch_bounds__(256) void ln_last_k(const float* __restrict__ h,
                                                 const float* __restrict__ w,
                                                 const float* __restrict__ bp,
                                                 float* __restrict__ out) {
  __shared__ float red[256];
  int b = blockIdx.x;
  const float* xr = h + ((long)b * GS + (GS - 1)) * GD;
  int tid = threadIdx.x;

  float s = 0.f;
  for (int i = tid; i < GD; i += 256) s += xr[i];
  red[tid] = s; __syncthreads();
  for (int o = 128; o > 0; o >>= 1) { if (tid < o) red[tid] += red[tid + o]; __syncthreads(); }
  float mu = red[0] / (float)GD;
  __syncthreads();

  float v = 0.f;
  for (int i = tid; i < GD; i += 256) { float d = xr[i] - mu; v += d * d; }
  red[tid] = v; __syncthreads();
  for (int o = 128; o > 0; o >>= 1) { if (tid < o) red[tid] += red[tid + o]; __syncthreads(); }
  float rstd = rsqrtf(red[0] / (float)GD + 1e-5f);

  float* orow = out + (long)b * GD;
  for (int i = tid; i < GD; i += 256)
    orow[i] = (xr[i] - mu) * rstd * w[i] + bp[i];
}

// ============================ WMMA GEMM ====================================
// C[M,N] = act(A[M,K] @ Bm[K,N] + bias + residual)
// A,Bm: f16 row-major. Block tile 128x64, 8 waves of 32x32, K-step 32.
// A staged global->LDS with GLOBAL_LOAD_ASYNC_TO_LDS_B128 (ASYNCcnt);
// B staged transposed so B fragments are contiguous 16B LDS reads.

template <bool RELU, bool RES, bool OUTH>
__global__ __launch_bounds__(256) void gemm_wmma_k(
    const _Float16* __restrict__ A, const _Float16* __restrict__ Bm,
    const float* __restrict__ bias, const float* __restrict__ res,
    void* __restrict__ Cout, int M, int N, int K) {
  __shared__ __align__(16) _Float16 As[128][40];   // stride 80B (16B multiple)
  __shared__ __align__(16) _Float16 Bt[64][40];    // B^T: [n][k]

  const int t    = threadIdx.x;
  const int lane = t & 31;
  const int wave = t >> 5;
  const int bm   = blockIdx.y * 128;
  const int bn   = blockIdx.x * 64;
  const int wm   = (wave >> 1) * 32;   // 0,32,64,96
  const int wn   = (wave & 1) * 32;    // 0,32

  const int la_row = t >> 1;           // A-tile: 16 halfs/thread
  const int la_col = (t & 1) * 16;
  const int lb_k   = t >> 3;           // B-tile: 8 halfs/thread
  const int lb_n   = (t & 7) * 8;

  const int lcol = lane & 15;
  const int hsel = lane >> 4;

  // LDS byte offset of this thread's A-tile chunk (flat addr low 32 bits ==
  // LDS offset per the CDNA5 aperture mapping).
  const unsigned ldsA = (unsigned)(unsigned long long)&As[la_row][la_col];
  const _Float16* gA0 = A + (long)(bm + la_row) * K + la_col;
  const _Float16* gB0 = Bm + (long)lb_k * N + bn + lb_n;

  v8f c0 = {}, c1 = {}, c2 = {}, c3 = {};
  union U { v16h v; uint4 u[2]; };

  for (int k0 = 0; k0 < K; k0 += 32) {
    // ---- stage A (128x32): async global->LDS, inst_offset applies to both
    // the LDS and global address, so one base + offset:16 covers the pair ----
    {
      unsigned long long ga = (unsigned long long)(gA0 + k0);
      asm volatile("global_load_async_to_lds_b128 %0, %1, off"
                   :: "v"(ldsA), "v"(ga) : "memory");
      asm volatile("global_load_async_to_lds_b128 %0, %1, off offset:16"
                   :: "v"(ldsA), "v"(ga) : "memory");
    }
    // ---- stage B^T (64x32) via registers (transpose scatter) ----
    {
      const _Float16* src = gB0 + (long)k0 * N;
      uint4 u = *(const uint4*)src;
      const _Float16* hv = (const _Float16*)&u;
#pragma unroll
      for (int i = 0; i < 8; ++i) Bt[lb_n + i][lb_k] = hv[i];
    }
    // prefetch next K-step tiles while we compute on this one
    if (k0 + 32 < K) {
      __builtin_prefetch(gA0 + k0 + 32, 0, 1);
      __builtin_prefetch(gB0 + (long)(k0 + 32) * N, 0, 1);
    }
    asm volatile("s_wait_asynccnt 0x0" ::: "memory");
    __syncthreads();

    // fragments per documented 16-bit layouts:
    // A lane: m=lane%16, elems 0-7 = K[half*8 .. +7], 8-15 = K[16+half*8 .. +7]
    // B lane: n=lane%16, elems 0-7 = K[half*8 .. +7], 8-15 = K[16+half*8 .. +7] (via B^T rows)
    U a0, a1, b0, b1;
    a0.u[0] = *(const uint4*)&As[wm + lcol][hsel * 8];
    a0.u[1] = *(const uint4*)&As[wm + lcol][16 + hsel * 8];
    a1.u[0] = *(const uint4*)&As[wm + 16 + lcol][hsel * 8];
    a1.u[1] = *(const uint4*)&As[wm + 16 + lcol][16 + hsel * 8];
    b0.u[0] = *(const uint4*)&Bt[wn + lcol][hsel * 8];
    b0.u[1] = *(const uint4*)&Bt[wn + lcol][16 + hsel * 8];
    b1.u[0] = *(const uint4*)&Bt[wn + 16 + lcol][hsel * 8];
    b1.u[1] = *(const uint4*)&Bt[wn + 16 + lcol][16 + hsel * 8];

    c0 = __builtin_amdgcn_wmma_f32_16x16x32_f16(false, a0.v, false, b0.v, (short)0, c0, false, false);
    c1 = __builtin_amdgcn_wmma_f32_16x16x32_f16(false, a0.v, false, b1.v, (short)0, c1, false, false);
    c2 = __builtin_amdgcn_wmma_f32_16x16x32_f16(false, a1.v, false, b0.v, (short)0, c2, false, false);
    c3 = __builtin_amdgcn_wmma_f32_16x16x32_f16(false, a1.v, false, b1.v, (short)0, c3, false, false);
    __syncthreads();
  }

  // epilogue; C layout: VGPR r -> row r + 8*(lane/16), col = lane%16
  v8f cc[4] = {c0, c1, c2, c3};
#pragma unroll
  for (int f = 0; f < 4; ++f) {
    int mb  = bm + wm + (f >> 1) * 16;
    int col = bn + wn + (f & 1) * 16 + lcol;
    float bv = bias ? bias[col] : 0.f;
#pragma unroll
    for (int r = 0; r < 8; ++r) {
      int row = mb + r + hsel * 8;
      if (row < M) {
        long idx = (long)row * N + col;
        float v = cc[f][r] + bv;
        if (RES)  v += res[idx];
        if (RELU) v = v > 0.f ? v : 0.f;
        if (OUTH) ((_Float16*)Cout)[idx] = (_Float16)v;
        else      ((float*)Cout)[idx] = v;
      }
    }
  }
}

// ============================ attention ====================================
// One wave per (b, head, 16-query tile). Flash-style online softmax.
// q/k layout: [B*S, D] (head hd at cols hd*64..+63); Vt: [(b*H+h)*HS+e, S].

__global__ __launch_bounds__(32) void attn_k(const _Float16* __restrict__ Q,
                                             const _Float16* __restrict__ Km,
                                             const _Float16* __restrict__ Vt,
                                             const int* __restrict__ amask,
                                             _Float16* __restrict__ out) {
  const int blk = blockIdx.x;
  const int qt  = blk & 63;                // S/16 tiles
  const int hd  = (blk >> 6) % GH;
  const int b   = blk / (64 * GH);

  const int lane = threadIdx.x;
  const int lcol = lane & 15;
  const int hsel = lane >> 4;

  __shared__ __align__(16) float    sc[16][33];
  __shared__ __align__(16) _Float16 pL[16][32];
  __shared__ float aL[16];
  __shared__ float sL[16];

  union U { v16h v; uint4 u[2]; };

  // Q fragments (16x64 split into two 16x32 A-frags)
  const long rowQ = ((long)b * GS + qt * 16 + lcol) * GD + hd * GHS;
  U aq0, aq1;
  aq0.u[0] = *(const uint4*)(Q + rowQ + hsel * 8);
  aq0.u[1] = *(const uint4*)(Q + rowQ + 16 + hsel * 8);
  aq1.u[0] = *(const uint4*)(Q + rowQ + 32 + hsel * 8);
  aq1.u[1] = *(const uint4*)(Q + rowQ + 48 + hsel * 8);

  v8f o[4] = {};                 // 16x64 output accumulator (4 N-tiles)
  float mrun = -3.0e38f, srun = 0.f;   // valid in lanes 0-15 (row = lane)
  const float scale = 0.03608439182435161f;  // D^-0.5 (reference scales by D)

  // Vt row base for this head: lane's head-dim column per N-tile
  const long vtBase = ((long)b * GH + hd) * GHS;

  for (int kb = 0; kb < GS; kb += 32) {
    // ---- scores for two 16-key tiles ----
#pragma unroll
    for (int j = 0; j < 2; ++j) {
      const long rowK = ((long)b * GS + kb + j * 16 + lcol) * GD + hd * GHS;
      U bk0, bk1;   // B-frag of K^T: lane n=key, elems = e contiguous in K row
      bk0.u[0] = *(const uint4*)(Km + rowK + hsel * 8);
      bk0.u[1] = *(const uint4*)(Km + rowK + 16 + hsel * 8);
      bk1.u[0] = *(const uint4*)(Km + rowK + 32 + hsel * 8);
      bk1.u[1] = *(const uint4*)(Km + rowK + 48 + hsel * 8);
      v8f s = {};
      s = __builtin_amdgcn_wmma_f32_16x16x32_f16(false, aq0.v, false, bk0.v, (short)0, s, false, false);
      s = __builtin_amdgcn_wmma_f32_16x16x32_f16(false, aq1.v, false, bk1.v, (short)0, s, false, false);
      const bool mok = amask[(long)b * GS + kb + j * 16 + lcol] != 0;
#pragma unroll
      for (int r = 0; r < 8; ++r)
        sc[r + hsel * 8][j * 16 + lcol] = mok ? s[r] * scale : -3.0e38f;
    }
    __syncthreads();

    // ---- online softmax (row-per-lane, lanes 0-15) ----
    if (lane < 16) {
      const int row = lane;
      float mloc = -3.0e38f;
#pragma unroll 8
      for (int c = 0; c < 32; ++c) mloc = fmaxf(mloc, sc[row][c]);
      float mnew  = fmaxf(mrun, mloc);
      float alpha = __expf(mrun - mnew);
      float ssum  = 0.f;
#pragma unroll 8
      for (int c = 0; c < 32; ++c) {
        float e = __expf(sc[row][c] - mnew);
        ssum += e;
        pL[row][c] = (_Float16)e;
      }
      srun = srun * alpha + ssum;
      mrun = mnew;
      aL[row] = alpha;
    }
    __syncthreads();

    // rescale accumulators
    float al[8];
#pragma unroll
    for (int r = 0; r < 8; ++r) al[r] = aL[r + hsel * 8];
#pragma unroll
    for (int r = 0; r < 8; ++r) {
      o[0][r] *= al[r]; o[1][r] *= al[r]; o[2][r] *= al[r]; o[3][r] *= al[r];
    }

    // P fragment (16 queries x 32 keys)
    U p;
    p.u[0] = *(const uint4*)&pL[lcol][hsel * 8];
    p.u[1] = *(const uint4*)&pL[lcol][16 + hsel * 8];

    // P @ V: 4 N-tiles of head dim; Vt rows give contiguous keys per column
#pragma unroll
    for (int tv = 0; tv < 4; ++tv) {
      const _Float16* vr = Vt + (vtBase + tv * 16 + lcol) * GS + kb + hsel * 8;
      U bv;
      bv.u[0] = *(const uint4*)(vr);
      bv.u[1] = *(const uint4*)(vr + 16);
      o[tv] = __builtin_amdgcn_wmma_f32_16x16x32_f16(false, p.v, false, bv.v, (short)0, o[tv], false, false);
    }
    __syncthreads();
  }

  if (lane < 16) sL[lane] = 1.0f / srun;
  __syncthreads();
  float inv[8];
#pragma unroll
  for (int r = 0; r < 8; ++r) inv[r] = sL[r + hsel * 8];
#pragma unroll
  for (int tv = 0; tv < 4; ++tv)
#pragma unroll
    for (int r = 0; r < 8; ++r) {
      int row = qt * 16 + r + hsel * 8;
      out[((long)b * GS + row) * GD + hd * GHS + tv * 16 + lcol] =
          (_Float16)(o[tv][r] * inv[r]);
    }
}

// ============================ classifier head ==============================

__global__ __launch_bounds__(256) void cls1_k(const float* __restrict__ last,
                                              const float* __restrict__ W,
                                              const float* __restrict__ bias,
                                              float* __restrict__ mid) {
  int idx = blockIdx.x * 256 + threadIdx.x;
  if (idx >= GB * GDF) return;
  int bI = idx / GDF, j = idx % GDF;
  float acc = bias[j];
  const float* lr = last + (long)bI * GD;
  for (int d = 0; d < GD; ++d) acc += lr[d] * W[(long)d * GDF + j];
  mid[idx] = acc > 0.f ? acc : 0.f;
}

__global__ __launch_bounds__(128) void cls2_k(const float* __restrict__ mid,
                                              const float* __restrict__ W,
                                              const float* __restrict__ bias,
                                              float* __restrict__ out) {
  int idx = threadIdx.x;
  if (idx >= GB * GNC) return;
  int bI = idx / GNC, c = idx % GNC;
  float acc = bias[c];
  const float* mr = mid + (long)bI * GDF;
  for (int j = 0; j < GDF; ++j) acc += mr[j] * W[(long)j * GNC + c];
  out[idx] = acc;
}

// ============================ launch =======================================

extern "C" void kernel_launch(void* const* d_in, const int* in_sizes, int n_in,
                              void* d_out, int out_size, void* d_ws, size_t ws_size,
                              hipStream_t stream) {
  const int*   x     = (const int*)d_in[0];
  const int*   amask = (const int*)d_in[1];
  const float* tok   = (const float*)d_in[2];
  const float* pos   = (const float*)d_in[3];
  const float* Wq    = (const float*)d_in[4];
  const float* Wk    = (const float*)d_in[5];
  const float* Wv    = (const float*)d_in[6];
  const float* Wo    = (const float*)d_in[7];
  const float* bo    = (const float*)d_in[8];
  const float* ln1w  = (const float*)d_in[9];
  const float* ln1b  = (const float*)d_in[10];
  const float* ln2w  = (const float*)d_in[11];
  const float* ln2b  = (const float*)d_in[12];
  const float* W1    = (const float*)d_in[13];
  const float* b1    = (const float*)d_in[14];
  const float* W2    = (const float*)d_in[15];
  const float* b2    = (const float*)d_in[16];
  const float* lnfw  = (const float*)d_in[17];
  const float* lnfb  = (const float*)d_in[18];
  const float* cW1   = (const float*)d_in[19];
  const float* cb1   = (const float*)d_in[20];
  const float* cW2   = (const float*)d_in[21];
  const float* cb2   = (const float*)d_in[22];

  const long T = (long)GB * GS;       // 8192 tokens

  size_t off = 0;
  auto alloc = [&](size_t bytes) -> void* {
    void* p = (char*)d_ws + off;
    off = (off + bytes + 255) & ~(size_t)255;
    return p;
  };
  float*    h      = (float*)   alloc(T * GD * 4);
  _Float16* xn     = (_Float16*)alloc(T * GD * 2);
  _Float16* qb     = (_Float16*)alloc(T * GD * 2);
  _Float16* kbuf   = (_Float16*)alloc(T * GD * 2);
  _Float16* vb     = (_Float16*)alloc(T * GD * 2);
  _Float16* vt     = (_Float16*)alloc(T * GD * 2);
  _Float16* attout = (_Float16*)alloc(T * GD * 2);
  _Float16* mid    = (_Float16*)alloc(T * GDF * 2);
  _Float16* wqh    = (_Float16*)alloc((long)GL * GD * GD * 2);
  _Float16* wkh    = (_Float16*)alloc((long)GL * GD * GD * 2);
  _Float16* wvh    = (_Float16*)alloc((long)GL * GD * GD * 2);
  _Float16* woh    = (_Float16*)alloc((long)GL * GD * GD * 2);
  _Float16* w1h    = (_Float16*)alloc((long)GL * GD * GDF * 2);
  _Float16* w2h    = (_Float16*)alloc((long)GL * GDF * GD * 2);
  float*    last   = (float*)   alloc((long)GB * GD * 4);
  float*    clsmid = (float*)   alloc((long)GB * GDF * 4);

  // ---- weight conversion / repack (recomputed every call: deterministic) ----
  const long nQKV = (long)GL * GD * GD;
  repack_qkv_k<<<4096, 256, 0, stream>>>(Wq, wqh, nQKV);
  repack_qkv_k<<<4096, 256, 0, stream>>>(Wk, wkh, nQKV);
  repack_qkv_k<<<4096, 256, 0, stream>>>(Wv, wvh, nQKV);
  cvt_k<<<4096, 256, 0, stream>>>(Wo, woh, nQKV);
  cvt_k<<<8192, 256, 0, stream>>>(W1, w1h, (long)GL * GD * GDF);
  cvt_k<<<8192, 256, 0, stream>>>(W2, w2h, (long)GL * GDF * GD);

  // ---- embedding ----
  embed_k<<<(int)T, 256, 0, stream>>>(x, tok, pos, h);

  dim3 gD (GD  / 64, (int)(T / 128));   // N=768
  dim3 gDF(GDF / 64, (int)(T / 128));   // N=3072

  for (int l = 0; l < GL; ++l) {
    const long wOff  = (long)l * GD * GD;
    const long w1Off = (long)l * GD * GDF;
    const long w2Off = (long)l * GDF * GD;

    ln_k<<<(int)T, 256, 0, stream>>>(h, ln1w + l * GD, ln1b + l * GD, xn);

    gemm_wmma_k<false, false, true><<<gD, 256, 0, stream>>>(
        xn, wqh + wOff, nullptr, nullptr, qb, (int)T, GD, GD);
    gemm_wmma_k<false, false, true><<<gD, 256, 0, stream>>>(
        xn, wkh + wOff, nullptr, nullptr, kbuf, (int)T, GD, GD);
    gemm_wmma_k<false, false, true><<<gD, 256, 0, stream>>>(
        xn, wvh + wOff, nullptr, nullptr, vb, (int)T, GD, GD);

    vtrans_k<<<4096, 256, 0, stream>>>(vb, vt);

    attn_k<<<GB * GH * (GS / 16), 32, 0, stream>>>(qb, kbuf, vt, amask, attout);

    gemm_wmma_k<false, true, false><<<gD, 256, 0, stream>>>(
        attout, woh + wOff, bo + l * GD, h, h, (int)T, GD, GD);

    ln_k<<<(int)T, 256, 0, stream>>>(h, ln2w + l * GD, ln2b + l * GD, xn);

    gemm_wmma_k<true, false, true><<<gDF, 256, 0, stream>>>(
        xn, w1h + w1Off, b1 + l * GDF, nullptr, mid, (int)T, GDF, GD);

    gemm_wmma_k<false, true, false><<<gD, 256, 0, stream>>>(
        mid, w2h + w2Off, b2 + l * GD, h, h, (int)T, GD, GDF);
  }

  // ---- final LN (last token only) + classifier head ----
  ln_last_k<<<GB, 256, 0, stream>>>(h, lnfw, lnfb, last);
  cls1_k<<<(GB * GDF + 255) / 256, 256, 0, stream>>>(last, cW1, cb1, clsmid);
  cls2_k<<<1, 128, 0, stream>>>(clsmid, cW2, cb2, (float*)d_out);
}
